// AttnGruDecoder_44727789420725
// MI455X (gfx1250) — matrix operand order repaired and checked by reading, not compiled, not run on hardware
//
#include <hip/hip_runtime.h>
#include <hip/hip_bf16.h>

// ---------------------------------------------------------------------------
// AttnGRU decoder for MI455X (gfx1250, wave32, WMMA + TDM).
// Phase 1: fp32->bf16 conversion of weights/inputs (L2-resident thereafter).
// Phase 2: sequential recurrence (1 WG, 32 waves) using v_wmma_f32_16x16x32_bf16,
//          scores/softmax/ctx kept in LDS.
// Phase 3: big prediction GEMM (1536 x 28996 x 768); B slabs staged into LDS by
//          the Tensor Data Mover (tensor_load_to_lds, TENSORcnt), double-buffered.
// ---------------------------------------------------------------------------

typedef __attribute__((ext_vector_type(16))) __bf16 v16bf;
typedef __attribute__((ext_vector_type(8)))  __bf16 v8bf;
typedef __attribute__((ext_vector_type(4)))  __bf16 v4bf;
typedef __attribute__((ext_vector_type(8)))  float  v8f;
typedef __attribute__((ext_vector_type(4)))  unsigned int v4u;
typedef __attribute__((ext_vector_type(4)))  int    v4i;
typedef __attribute__((ext_vector_type(8)))  int    v8i;

#define B_   32
#define T_   48
#define D_   768
#define H_   768
#define S_   128
#define V_   28996

#if defined(__has_builtin)
#if __has_builtin(__builtin_amdgcn_tensor_load_to_lds) && \
    __has_builtin(__builtin_amdgcn_s_wait_tensorcnt)
#define USE_TDM 1
#endif
#endif
#ifndef USE_TDM
#define USE_TDM 0
#endif

static __device__ __forceinline__ float sigmoidf_(float x) {
    return 1.0f / (1.0f + __expf(-x));
}

#if USE_TDM
// Issue a TDM 2D tile load: tile_dim0=32 elements (K, contiguous within a
// W_pred row), tile_dim1=64 rows (N), row stride 768 elements, 2-byte data.
// Rows >= nrows read as zero (handles the ragged V edge).  lds_off is the
// byte offset in the workgroup LDS segment (Bs is the only static __shared__
// allocation in the kernel, so it sits at group-segment offset 0).
static __device__ __forceinline__ void tdm_load_slab(unsigned lds_off,
                                                     const __bf16* gsrc,
                                                     int nrows) {
    unsigned long long ga = (unsigned long long)(uintptr_t)gsrc;
    v4u g0;
    g0[0] = 1u;                                   // count=1 valid, user mode
    g0[1] = lds_off;                              // lds_addr (bytes)
    g0[2] = (unsigned)(ga & 0xffffffffu);         // global_addr[31:0]
    g0[3] = (unsigned)((ga >> 32) & 0x01ffffffu)  // global_addr[56:32]
            | (2u << 30);                         // type = 2 ("image")
    v8i g1;
    g1[0] = (int)(1u << 16);              // data_size=1 (2B), no mcast/flags
    g1[1] = (int)(768u << 16);            // tensor_dim0 = 768 (elements)
    g1[2] = (int)((unsigned)nrows << 16); // tensor_dim1 = valid rows
    g1[3] = (int)(32u << 16);             // tile_dim0 = 32
    g1[4] = 64;                           // tile_dim1 = 64, tile_dim2 = 0
    g1[5] = 768;                          // tensor_dim0_stride[31:0] = 768
    g1[6] = 0;                            // stride hi / dim1_stride lo
    g1[7] = 0;                            // dim1_stride hi (unused, 2D)
    v4i z = {0, 0, 0, 0};
#if defined(__clang_major__) && __clang_major__ >= 23
    v8i z8 = {0, 0, 0, 0, 0, 0, 0, 0};
    __builtin_amdgcn_tensor_load_to_lds(g0, g1, z, z, z8, 0);
#else
    __builtin_amdgcn_tensor_load_to_lds(g0, g1, z, z, 0);
#endif
}
#endif  // USE_TDM

// A fragment: 16x32 (MxK) bf16, contiguous K order (matches load_B_frag):
//   lanes 0-15 (m=lane):     halves 0-15 -> K=k0..k0+15
//   lanes 16-31 (m=lane-16): halves 0-15 -> K=k0+16..k0+31
static __device__ __forceinline__ v16bf load_A_frag(const __bf16* A, int lda,
                                                    int m_base, int k0, int lane) {
    const __bf16* row = A + (size_t)(m_base + (lane & 15)) * (size_t)lda;
    int d = (lane >> 4) << 4;  // 0 or 16
    union { v16bf v; v8bf h[2]; } u;
    u.h[0] = *(const v8bf*)(row + k0 + d);
    u.h[1] = *(const v8bf*)(row + k0 + 8 + d);
    return u.v;
}

// B fragment: 32x16 (KxN) bf16. W is row-major (N x K): B(k,n) = W[n*ldb + k].
//   lanes 0-15:  n=lane,     K = k0..k0+15 (contiguous)
//   lanes 16-31: n=lane-16,  K = k0+16..k0+31
static __device__ __forceinline__ v16bf load_B_frag(const __bf16* W, int ldb,
                                                    int n_base, int k0, int lane) {
    const __bf16* row = W + (size_t)(n_base + (lane & 15)) * (size_t)ldb;
    return *(const v16bf*)(row + k0 + ((lane >> 4) << 4));
}

// Generic small GEMM over tiles, nwaves waves cooperating.
//   C(M x N) = [A0 | A1](M x K) * W(N x K)^T + bias   (A split at column kSplit)
//   If CoutBf != nullptr: store tanh(C) as bf16; else store fp32.
static __device__ void wmma_gemm_ws(const __bf16* A0, const __bf16* A1, int kSplit,
                                    int lda, const __bf16* W, int ldb,
                                    const float* bias, int M, int Ntiles, int K,
                                    float* Cout, int ldc,
                                    __bf16* CoutBf, int ldcb,
                                    int wave, int nwaves, int lane) {
    const int Mtiles  = M >> 4;
    const int ksteps  = K >> 5;
    const int ksteps0 = ((kSplit >> 5) < ksteps) ? (kSplit >> 5) : ksteps;
    const int total   = Mtiles * Ntiles;
    for (int tile = wave; tile < total; tile += nwaves) {
        const int mt = tile % Mtiles;
        const int nt = tile / Mtiles;
        const int m_base = mt << 4;
        const int n_base = nt << 4;
        v8f acc = {};
#pragma unroll 2
        for (int ks = 0; ks < ksteps0; ++ks) {
            const int k0 = ks << 5;
            v16bf af = load_A_frag(A0, lda, m_base, k0, lane);
            v16bf bf = load_B_frag(W, ldb, n_base, k0, lane);
            acc = __builtin_amdgcn_wmma_f32_16x16x32_bf16(
                false, af, false, bf, (short)0, acc, false, false);
        }
#pragma unroll 2
        for (int ks = ksteps0; ks < ksteps; ++ks) {
            const int k0 = ks << 5;
            v16bf af = load_A_frag(A1, lda, m_base, k0 - kSplit, lane);
            v16bf bf = load_B_frag(W, ldb, n_base, k0, lane);
            acc = __builtin_amdgcn_wmma_f32_16x16x32_bf16(
                false, af, false, bf, (short)0, acc, false, false);
        }
        const int n  = n_base + (lane & 15);
        const float bv = bias ? bias[n] : 0.0f;
        const int mo = (lane >> 4) << 3;  // 0 or 8
#pragma unroll
        for (int r = 0; r < 8; ++r) {
            const int m = m_base + r + mo;
            const float v = acc[r] + bv;
            if (CoutBf) CoutBf[(size_t)m * ldcb + n] = (__bf16)tanhf(v);
            else        Cout  [(size_t)m * ldc  + n] = v;
        }
    }
}

// ---------------------------------------------------------------------------
// Phase 1: fp32 -> bf16 conversion (vectorized x4)
// ---------------------------------------------------------------------------
__global__ void cvt_f32_bf16_kernel(const float* __restrict__ src,
                                    __bf16* __restrict__ dst, int n4) {
    int i = blockIdx.x * blockDim.x + threadIdx.x;
    if (i < n4) {
        float4 v = ((const float4*)src)[i];
        v4bf o = { (__bf16)v.x, (__bf16)v.y, (__bf16)v.z, (__bf16)v.w };
        ((v4bf*)dst)[i] = o;
    }
}

// ---------------------------------------------------------------------------
// Phase 2: sequential recurrence, single workgroup (32 waves).
// Dynamic LDS: scores (B*S fp32 = 16KB) + ctx (B*H bf16 = 48KB) = 64KB.
// ---------------------------------------------------------------------------
__global__ __launch_bounds__(1024, 1)
void recurrence_kernel(const __bf16* __restrict__ x_bf,     // (B,T,D) bf16
                       const float*  __restrict__ enc,       // (B,S,H) fp32
                       const __bf16* __restrict__ Wih_bf,    // (3H, D)
                       const __bf16* __restrict__ Whh_bf,    // (3H, H)
                       const float*  __restrict__ bih,
                       const float*  __restrict__ bhh,
                       const __bf16* __restrict__ Watt_bf,   // (H, H)
                       const float*  __restrict__ batt,
                       const __bf16* __restrict__ Wctx_bf,   // (H, 2H)
                       const float*  __restrict__ bctx,
                       __bf16* __restrict__ h_cur,           // (B,H) state (pre-init)
                       __bf16* __restrict__ h_gru,           // (T,B,H) GRU outputs
                       float*  __restrict__ Gi,              // (B,3H)
                       float*  __restrict__ Gh,              // (B,3H)
                       float*  __restrict__ a_att) {         // (B,H)
    extern __shared__ char smem[];
    float*  scores_s = (float*)smem;                  // 32*128 fp32
    __bf16* ctx_s    = (__bf16*)(smem + B_ * S_ * 4); // 32*768 bf16

    const int tid  = threadIdx.x;
    const int lane = tid & 31;
    const int wave = tid >> 5;

    for (int t = 0; t < T_; ++t) {
        // --- GRU gate GEMMs: Gi = x_t @ Wih^T + bih ; Gh = h @ Whh^T + bhh ---
        const __bf16* xt = x_bf + (size_t)t * D_;  // row b at xt + b*(T*D)
        wmma_gemm_ws(xt, nullptr, 1 << 30, T_ * D_, Wih_bf, D_, bih,
                     B_, (3 * H_) / 16, D_, Gi, 3 * H_, nullptr, 0, wave, 32, lane);
        wmma_gemm_ws(h_cur, nullptr, 1 << 30, H_, Whh_bf, H_, bhh,
                     B_, (3 * H_) / 16, H_, Gh, 3 * H_, nullptr, 0, wave, 32, lane);
        __threadfence_block(); __syncthreads();

        // --- gates elementwise -> h_gru[t] ---
        __bf16* hg = h_gru + (size_t)t * B_ * H_;
        for (int j = tid; j < B_ * H_; j += 1024) {
            const int b = j / H_, k = j - b * H_;
            const size_t g0 = (size_t)b * (3 * H_) + k;
            const float r = sigmoidf_(Gi[g0] + Gh[g0]);
            const float z = sigmoidf_(Gi[g0 + H_] + Gh[g0 + H_]);
            const float n = tanhf(Gi[g0 + 2 * H_] + r * Gh[g0 + 2 * H_]);
            const float hp = (float)h_cur[j];
            hg[j] = (__bf16)((1.0f - z) * n + z * hp);
        }
        __threadfence_block(); __syncthreads();

        // --- a = h_gru @ Watt^T + batt ---
        wmma_gemm_ws(hg, nullptr, 1 << 30, H_, Watt_bf, H_, batt,
                     B_, H_ / 16, H_, a_att, H_, nullptr, 0, wave, 32, lane);
        __threadfence_block(); __syncthreads();

        // --- scores[b,s] = dot(a[b], enc[b,s]) : one wave per (b,s) dot ---
        for (int idx = wave; idx < B_ * S_; idx += 32) {
            const int b = idx >> 7, s = idx & (S_ - 1);
            const float* ar = a_att + (size_t)b * H_;
            const float* er = enc + ((size_t)b * S_ + s) * H_;
            float p = 0.0f;
            for (int k = lane; k < H_; k += 32) p += ar[k] * er[k];
#pragma unroll
            for (int off = 16; off; off >>= 1) p += __shfl_xor(p, off, 32);
            if (lane == 0) scores_s[idx] = p;
        }
        __syncthreads();

        // --- softmax over S per batch row: wave w owns row b=w ---
        {
            const int b = wave;
            float v[S_ / 32];
            float mx = -1e30f;
#pragma unroll
            for (int i = 0; i < S_ / 32; ++i) {
                v[i] = scores_s[b * S_ + lane + 32 * i];
                mx = fmaxf(mx, v[i]);
            }
#pragma unroll
            for (int off = 16; off; off >>= 1) mx = fmaxf(mx, __shfl_xor(mx, off, 32));
            float sum = 0.0f;
#pragma unroll
            for (int i = 0; i < S_ / 32; ++i) { v[i] = __expf(v[i] - mx); sum += v[i]; }
#pragma unroll
            for (int off = 16; off; off >>= 1) sum += __shfl_xor(sum, off, 32);
            const float inv = 1.0f / sum;
#pragma unroll
            for (int i = 0; i < S_ / 32; ++i) scores_s[b * S_ + lane + 32 * i] = v[i] * inv;
        }
        __syncthreads();

        // --- ctx[b,k] = sum_s p[b,s] * enc[b,s,k] : wave w owns b=w ---
        {
            const int b = wave;
            const float* er = enc + (size_t)b * S_ * H_;
            const float* pr = scores_s + b * S_;
            for (int k = lane; k < H_; k += 32) {
                float c = 0.0f;
                for (int s = 0; s < S_; ++s) c += pr[s] * er[(size_t)s * H_ + k];
                ctx_s[(size_t)b * H_ + k] = (__bf16)c;
            }
        }
        __syncthreads();

        // --- h_new = tanh([ctx | h_gru] @ Wctx^T + bctx) -> h_cur (bf16) ---
        wmma_gemm_ws(ctx_s, hg, H_, H_, Wctx_bf, 2 * H_, bctx,
                     B_, H_ / 16, 2 * H_, nullptr, 0, h_cur, H_, wave, 32, lane);
        __threadfence_block(); __syncthreads();
    }
}

// ---------------------------------------------------------------------------
// Phase 3: prediction GEMM.  out[b,t,v] = h_gru[t,b,:] @ W_pred[v,:] + b_pred[v]
//   M = T*B = 1536 rows (row m -> t=m>>5, b=m&31), N = V = 28996, K = 768.
//   Block = 8 waves sharing a 64-column B slab, double-buffered through LDS.
//   TDM path: wave 0 issues tensor_load_to_lds per K-step (ASYNC DMA) and
//   synchronizes with s_wait_tensorcnt + workgroup barrier.
// ---------------------------------------------------------------------------
__global__ __launch_bounds__(256, 4)
void pred_gemm_kernel(const __bf16* __restrict__ hgru,     // (T*B, H) bf16
                      const __bf16* __restrict__ Wpred_bf, // (V, H) bf16
                      const float*  __restrict__ bpred,
                      float* __restrict__ out) {
    __shared__ __align__(128) __bf16 Bs[2][64 * 32];  // 2 x 4 KB, LDS offset 0

    const int tid  = threadIdx.x;
    const int lane = tid & 31;
    const int wave = tid >> 5;
    const int n0     = blockIdx.x * 64;             // N slab base
    const int m_base = (blockIdx.y * 8 + wave) << 4;
    const int KSTEPS = H_ >> 5;                     // 24

    v8f acc[4] = {};

#if USE_TDM
    const int nrows = (V_ - n0 > 64) ? 64 : (V_ - n0);
    const __bf16* slab = Wpred_bf + (size_t)n0 * H_;

    if (wave == 0) {
        tdm_load_slab(0u, slab, nrows);                 // K-slab 0 -> buffer 0
        __builtin_amdgcn_s_wait_tensorcnt(0);
    }
    __syncthreads();

    for (int ks = 0; ks < KSTEPS; ++ks) {
        const int cur = ks & 1;
        const int k0  = ks << 5;
        // DMA next K-slab into the other buffer while we compute this one.
        // Safe: all readers of buf[cur^1] passed the previous barrier.
        if (ks < KSTEPS - 1 && wave == 0)
            tdm_load_slab((unsigned)((cur ^ 1) * (64 * 32 * 2)),
                          slab + k0 + 32, nrows);

        v16bf af = load_A_frag(hgru, H_, m_base, k0, lane);
#pragma unroll
        for (int sub = 0; sub < 4; ++sub) {
            v16bf bf = load_B_frag(&Bs[cur][0], 32, sub << 4, 0, lane);
            acc[sub] = __builtin_amdgcn_wmma_f32_16x16x32_bf16(
                false, af, false, bf, (short)0, acc[sub], false, false);
        }
        if (wave == 0) __builtin_amdgcn_s_wait_tensorcnt(0);
        __syncthreads();  // slab for ks+1 complete; reads of buf[cur] done
    }
#else
    // Fallback: register-staged double buffer (each thread owns 8 halves)
    const int snn = tid >> 2;
    const int sch = (tid & 3) << 3;
    const int sn  = n0 + snn;
    const bool svalid = (sn < V_);
    const __bf16* srow = Wpred_bf + (size_t)sn * H_;
    {
        v8bf v = {};
        if (svalid) v = *(const v8bf*)(srow + sch);
        *(v8bf*)(&Bs[0][snn * 32 + sch]) = v;
    }
    __syncthreads();
#pragma unroll 2
    for (int ks = 0; ks < KSTEPS; ++ks) {
        const int cur = ks & 1;
        const int k0  = ks << 5;
        v8bf nxt = {};
        if (ks < KSTEPS - 1 && svalid) {
            nxt = *(const v8bf*)(srow + k0 + 32 + sch);
            if (ks < KSTEPS - 2)
                __builtin_prefetch(srow + k0 + 64 + sch, 0, 1);
        }
        v16bf af = load_A_frag(hgru, H_, m_base, k0, lane);
#pragma unroll
        for (int sub = 0; sub < 4; ++sub) {
            v16bf bf = load_B_frag(&Bs[cur][0], 32, sub << 4, 0, lane);
            acc[sub] = __builtin_amdgcn_wmma_f32_16x16x32_bf16(
                false, af, false, bf, (short)0, acc[sub], false, false);
        }
        if (ks < KSTEPS - 1)
            *(v8bf*)(&Bs[cur ^ 1][snn * 32 + sch]) = nxt;
        __syncthreads();
    }
#endif

    // epilogue: out[(b*T + t)*V + n]
    const int mo = (lane >> 4) << 3;
#pragma unroll
    for (int sub = 0; sub < 4; ++sub) {
        const int n = n0 + (sub << 4) + (lane & 15);
        if (n < V_) {
            const float bv = bpred[n];
#pragma unroll
            for (int r = 0; r < 8; ++r) {
                const int m = m_base + r + mo;
                const int t = m >> 5, b = m & 31;
                out[((size_t)b * T_ + t) * V_ + n] = acc[sub][r] + bv;
            }
        }
    }
}

// ---------------------------------------------------------------------------
// Host launcher
// ---------------------------------------------------------------------------
extern "C" void kernel_launch(void* const* d_in, const int* in_sizes, int n_in,
                              void* d_out, int out_size, void* d_ws, size_t ws_size,
                              hipStream_t stream) {
    const float* x     = (const float*)d_in[0];
    const float* h0    = (const float*)d_in[1];
    const float* enc   = (const float*)d_in[2];
    const float* Wih   = (const float*)d_in[3];
    const float* Whh   = (const float*)d_in[4];
    const float* bih   = (const float*)d_in[5];
    const float* bhh   = (const float*)d_in[6];
    const float* Wpred = (const float*)d_in[7];
    const float* bpred = (const float*)d_in[8];
    const float* Watt  = (const float*)d_in[9];
    const float* batt  = (const float*)d_in[10];
    const float* Wctx  = (const float*)d_in[11];
    const float* bctx  = (const float*)d_in[12];
    float* out = (float*)d_out;

    // workspace carve-out (256B aligned)
    char* wsb = (char*)d_ws;
    size_t off = 0;
    auto carve = [&](size_t bytes) -> void* {
        void* p = wsb + off;
        off = (off + bytes + 255) & ~(size_t)255;
        return p;
    };
    __bf16* Wpred_bf = (__bf16*)carve((size_t)V_ * H_ * 2);
    __bf16* Wih_bf   = (__bf16*)carve((size_t)3 * H_ * D_ * 2);
    __bf16* Whh_bf   = (__bf16*)carve((size_t)3 * H_ * H_ * 2);
    __bf16* Watt_bf  = (__bf16*)carve((size_t)H_ * H_ * 2);
    __bf16* Wctx_bf  = (__bf16*)carve((size_t)H_ * 2 * H_ * 2);
    __bf16* x_bf     = (__bf16*)carve((size_t)B_ * T_ * D_ * 2);
    __bf16* h_gru    = (__bf16*)carve((size_t)T_ * B_ * H_ * 2);
    __bf16* h_cur    = (__bf16*)carve((size_t)B_ * H_ * 2);
    float*  Gi       = (float*) carve((size_t)B_ * 3 * H_ * 4);
    float*  Gh       = (float*) carve((size_t)B_ * 3 * H_ * 4);
    float*  a_att    = (float*) carve((size_t)B_ * H_ * 4);

    // Phase 1: conversions
    auto cvt = [&](const float* s, __bf16* d, size_t n) {
        int n4 = (int)(n >> 2);
        cvt_f32_bf16_kernel<<<(n4 + 255) / 256, 256, 0, stream>>>(s, d, n4);
    };
    cvt(Wpred, Wpred_bf, (size_t)V_ * H_);
    cvt(Wih,   Wih_bf,   (size_t)3 * H_ * D_);
    cvt(Whh,   Whh_bf,   (size_t)3 * H_ * H_);
    cvt(Watt,  Watt_bf,  (size_t)H_ * H_);
    cvt(Wctx,  Wctx_bf,  (size_t)H_ * 2 * H_);
    cvt(x,     x_bf,     (size_t)B_ * T_ * D_);
    cvt(h0,    h_cur,    (size_t)B_ * H_);

    // Phase 2: sequential recurrence (one workgroup, 32 waves, 64KB dynamic LDS)
    const size_t smem = (size_t)B_ * S_ * 4 + (size_t)B_ * H_ * 2;  // 64 KB
    recurrence_kernel<<<1, 1024, smem, stream>>>(
        x_bf, enc, Wih_bf, Whh_bf, bih, bhh, Watt_bf, batt, Wctx_bf, bctx,
        h_cur, h_gru, Gi, Gh, a_att);

    // Phase 3: prediction GEMM
    const int ngroups = (V_ + 63) / 64;       // 454 slabs of 64 columns
    const int mgroups = (T_ * B_) / (16 * 8); // 12 groups of 8 M-tiles
    pred_gemm_kernel<<<dim3(ngroups, mgroups), 256, 0, stream>>>(
        h_gru, Wpred_bf, bpred, out);
}
